// GNNQNetwork_51101520888522
// MI455X (gfx1250) — matrix operand order repaired and checked by reading, not compiled, not run on hardware
//
#include <hip/hip_runtime.h>
#include <hip/hip_bf16.h>
#include <stdint.h>

#define HID 128

typedef __attribute__((ext_vector_type(16))) __bf16 v16bf;
typedef __attribute__((ext_vector_type(8)))  float  v8f;
typedef __attribute__((address_space(3))) const void* lds_cvp;

// ---------------------------------------------------------------------------
// WMMA helpers (gfx1250, wave32). D = A(16x32 bf16) x B(32x16 bf16) + C(f32)
// ---------------------------------------------------------------------------
__device__ __forceinline__ v8f wmma_bf16(v16bf a, v16bf b, v8f c) {
  return __builtin_amdgcn_wmma_f32_16x16x32_bf16(
      false, a, false, b, (short)0, c, false, false);
}

// A-matrix 16x32 bf16 from LDS fp32 rows (ISA lane layout: lanes 0-15 M=lane,
// K {0..7,16..23}; lanes 16-31 M=lane-16, K {8..15,24..31}).
__device__ __forceinline__ v16bf load_A(const float* lds, int stride, int k0) {
  int lane = threadIdx.x & 31;
  const float* p = lds + (lane & 15) * stride + k0 + ((lane < 16) ? 0 : 8);
  v16bf a;
#pragma unroll
  for (int j = 0; j < 8; ++j) {
    int k = (j < 4) ? (2 * j) : (8 + 2 * j);
    a[2 * j]     = (__bf16)p[k];
    a[2 * j + 1] = (__bf16)p[k + 1];
  }
  return a;
}

// B-tile load from pre-swizzled bf16 weights: tile t, lane -> 16 contiguous
// bf16 (32B). Pack kernel already applied W^T access, lane layout, K padding.
__device__ __forceinline__ v16bf load_Bp(const __bf16* __restrict__ P, int t) {
  int lane = threadIdx.x & 31;
  return *(const v16bf*)(P + ((long)t * 32 + lane) * 16);
}

// Async copy of one 512B feature row (128 f32) global -> LDS: 32 lanes x 16B.
__device__ __forceinline__ void async_row512(const float* __restrict__ g,
                                             const float* l) {
  int lane = threadIdx.x & 31;
  unsigned loff = (unsigned)(size_t)(lds_cvp)(l + lane * 4);
  unsigned long long ga = (unsigned long long)(size_t)(g + lane * 4);
  asm volatile("global_load_async_to_lds_b128 %0, %1, off"
               :: "v"(loff), "v"(ga) : "memory");
}
__device__ __forceinline__ void wait_async0() {
  asm volatile("s_wait_asynccnt 0x0" ::: "memory");
}

// ---------------------------------------------------------------------------
// Weight pre-pack: P[t*512 + lane*16 + j] = bf16(W[(ct*16+lane%16)*ld + k]),
// k = kc*32 + (lane<16?0:8) + (j<8?j:j+8), zero-padded at k>=kmax.  t = kc*ctn + ct.
// ---------------------------------------------------------------------------
__global__ void k_pack(const float* __restrict__ W, __bf16* __restrict__ P,
                       int ld, int kmax, int ctn, int total) {
  int idx = blockIdx.x * 256 + threadIdx.x;
  if (idx >= total) return;
  int j = idx & 15, lane = (idx >> 4) & 31, t = idx >> 9;
  int kc = t / ctn, ct = t - kc * ctn;
  int col = ct * 16 + (lane & 15);
  int k = kc * 32 + ((lane < 16) ? 0 : 8) + ((j < 8) ? j : j + 8);
  P[idx] = (k < kmax) ? (__bf16)W[(long)col * ld + k] : (__bf16)0.0f;
}

// ---------------------------------------------------------------------------
// Kernel 1: node projection  h = relu(x @ Wnp^T + b)
// ---------------------------------------------------------------------------
__global__ void k_node_proj(const float* __restrict__ x,
                            const __bf16* __restrict__ Pw,
                            const float* __restrict__ b,
                            float* __restrict__ h, int N) {
  __shared__ float sx[2][16][36];
  int lane = threadIdx.x & 31, wv = threadIdx.x >> 5;
  int nbase = (blockIdx.x * 2 + wv) * 16;
  for (int i = lane; i < 16 * 32; i += 32) {
    int r = i >> 5, c = i & 31;
    int node = nbase + r;
    sx[wv][r][c] = (node < N) ? x[(long)node * 32 + c] : 0.0f;
  }
  __syncthreads();
  v16bf a = load_A(&sx[wv][0][0], 36, 0);
  int cl = lane & 15, hi = (lane < 16) ? 0 : 8;
  bool full = (nbase + 16 <= N);
#pragma unroll
  for (int ct = 0; ct < 8; ++ct) {
    v8f acc = {};
    acc = wmma_bf16(a, load_Bp(Pw, ct), acc);
    int col = ct * 16 + cl;
    float bias = b[col];
    if (full) {
#pragma unroll
      for (int r = 0; r < 8; ++r) {
        float v = acc[r] + bias;
        h[(long)(nbase + hi + r) * HID + col] = v > 0.f ? v : 0.f;
      }
    } else {
#pragma unroll
      for (int r = 0; r < 8; ++r) {
        int node = nbase + hi + r;
        if (node < N) {
          float v = acc[r] + bias;
          h[(long)node * HID + col] = v > 0.f ? v : 0.f;
        }
      }
    }
  }
}

// ---------------------------------------------------------------------------
// Kernel 2: attention scores. [hs|hd|ea] (272) -> 64 (lrelu) -> 1.
// ---------------------------------------------------------------------------
__global__ void k_attn(const float* __restrict__ h, const float* __restrict__ ea,
                       const int* __restrict__ src, const int* __restrict__ dst,
                       const __bf16* __restrict__ Pw1, const float* __restrict__ ab1,
                       const float* __restrict__ aw2, const float* __restrict__ ab2,
                       float* __restrict__ scores, unsigned* __restrict__ maxbits,
                       int E) {
  __shared__ float sa[2][16][292];
  int lane = threadIdx.x & 31, wv = threadIdx.x >> 5;
  int ebase = (blockIdx.x * 2 + wv) * 16;
  float* row0 = &sa[wv][0][0];
  for (int i = 0; i < 16; ++i) {
    int e = ebase + i; if (e >= E) e = E - 1;
    int s = src[e], d = dst[e];
    float* row = row0 + i * 292;
    async_row512(h + (long)s * HID, row);          // K 0..127
    async_row512(h + (long)d * HID, row + 128);    // K 128..255
    if (lane < 16) row[256 + lane] = ea[(long)e * 16 + lane];
    else           row[256 + lane] = 0.0f;         // zero pad K 272..287
  }
  wait_async0();
  __syncthreads();
  v8f acc[4] = {};
  for (int kc = 0; kc < 9; ++kc) {
    v16bf a = load_A(row0, 292, kc * 32);
#pragma unroll
    for (int ct = 0; ct < 4; ++ct)
      acc[ct] = wmma_bf16(a, load_Bp(Pw1, kc * 4 + ct), acc[ct]);
  }
  int cl = lane & 15, hi = (lane < 16) ? 0 : 8;
  float part[8];
#pragma unroll
  for (int r = 0; r < 8; ++r) part[r] = 0.f;
#pragma unroll
  for (int ct = 0; ct < 4; ++ct) {
    int j = ct * 16 + cl;
    float b1 = ab1[j], w2 = aw2[j];
#pragma unroll
    for (int r = 0; r < 8; ++r) {
      float v = acc[ct][r] + b1;
      v = v > 0.f ? v : 0.2f * v;
      part[r] += v * w2;
    }
  }
#pragma unroll
  for (int r = 0; r < 8; ++r) {
    part[r] += __shfl_xor(part[r], 1);
    part[r] += __shfl_xor(part[r], 2);
    part[r] += __shfl_xor(part[r], 4);
    part[r] += __shfl_xor(part[r], 8);
  }
  float b2 = ab2[0];
  float lmax = -3.0e38f;
  if (cl == 0) {
    if (ebase + 16 <= E) {
#pragma unroll
      for (int r = 0; r < 8; ++r) {
        float sc = part[r] + b2;
        scores[ebase + hi + r] = sc;
        lmax = fmaxf(lmax, sc);
      }
    } else {
#pragma unroll
      for (int r = 0; r < 8; ++r) {
        int e = ebase + hi + r;
        if (e < E) {
          float sc = part[r] + b2;
          scores[e] = sc;
          lmax = fmaxf(lmax, sc);
        }
      }
    }
  }
  lmax = fmaxf(lmax, __shfl_xor(lmax, 16));
  if (lane == 0) {
    unsigned u = __float_as_uint(lmax);
    u = (u & 0x80000000u) ? ~u : (u | 0x80000000u);
    atomicMax(maxbits, u);
  }
}

// ---------------------------------------------------------------------------
// Kernel 3: exp(score - max) + per-dst denominator (f32 atomics, L2 resident)
// ---------------------------------------------------------------------------
__global__ void k_expsum(const float* __restrict__ scores, const int* __restrict__ dst,
                         const unsigned* __restrict__ maxbits,
                         float* __restrict__ expsc, float* __restrict__ sum_exp, int E) {
  unsigned u = *maxbits;
  unsigned ob = (u & 0x80000000u) ? (u ^ 0x80000000u) : ~u;
  float mx = __uint_as_float(ob);
  int t = blockIdx.x * blockDim.x + threadIdx.x;
  if (t < E) {
    float e = __expf(scores[t] - mx);
    expsc[t] = e;
    atomicAdd(&sum_exp[dst[t]], e);
  }
}

// ---------------------------------------------------------------------------
// Kernel 4: message MLP + weighted atomic scatter into agg[dst]
// ---------------------------------------------------------------------------
__global__ void k_msg(const float* __restrict__ h, const float* __restrict__ ea,
                      const int* __restrict__ src, const int* __restrict__ dst,
                      const __bf16* __restrict__ Pw1, const float* __restrict__ mb1,
                      const __bf16* __restrict__ Pw2, const float* __restrict__ mb2,
                      const float* __restrict__ expsc, const float* __restrict__ sum_exp,
                      float* __restrict__ agg, int E) {
  __shared__ float sin_[2][16][164];
  __shared__ float sm1[2][16][132];
  int lane = threadIdx.x & 31, wv = threadIdx.x >> 5;
  int ebase = (blockIdx.x * 2 + wv) * 16;
  float* in0 = &sin_[wv][0][0];
  float* m10 = &sm1[wv][0][0];
  for (int i = 0; i < 16; ++i) {
    int e = ebase + i; if (e >= E) e = E - 1;
    float* row = in0 + i * 164;
    async_row512(h + (long)src[e] * HID, row);
    if (lane < 16) row[128 + lane] = ea[(long)e * 16 + lane];
    else           row[128 + lane] = 0.0f;        // zero pad K 144..159
  }
  wait_async0();
  __syncthreads();
  int cl = lane & 15, hi = (lane < 16) ? 0 : 8;
  bool full = (ebase + 16 <= E);
  // GEMM1: [16x144] @ [144x128]
  v8f acc1[8] = {};
  for (int kc = 0; kc < 5; ++kc) {
    v16bf a = load_A(in0, 164, kc * 32);
#pragma unroll
    for (int ct = 0; ct < 8; ++ct)
      acc1[ct] = wmma_bf16(a, load_Bp(Pw1, kc * 8 + ct), acc1[ct]);
  }
#pragma unroll
  for (int ct = 0; ct < 8; ++ct) {
    int col = ct * 16 + cl;
    float b1 = mb1[col];
#pragma unroll
    for (int r = 0; r < 8; ++r) {
      float v = acc1[ct][r] + b1;
      m10[(hi + r) * 132 + col] = v > 0.f ? v : 0.f;
    }
  }
  __syncthreads();
  // attention weights per edge row
  float wrow[8];
#pragma unroll
  for (int r = 0; r < 8; ++r) {
    int e = ebase + hi + r;
    if (full || e < E) wrow[r] = expsc[e] / (sum_exp[dst[e]] + 1e-6f);
    else               wrow[r] = 0.0f;
  }
  // GEMM2: [16x128] @ [128x128]
  v8f acc2[8] = {};
  for (int kc = 0; kc < 4; ++kc) {
    v16bf a = load_A(m10, 132, kc * 32);
#pragma unroll
    for (int ct = 0; ct < 8; ++ct)
      acc2[ct] = wmma_bf16(a, load_Bp(Pw2, kc * 8 + ct), acc2[ct]);
  }
  if (full) {
#pragma unroll
    for (int ct = 0; ct < 8; ++ct) {
      int col = ct * 16 + cl;
      float b2 = mb2[col];
#pragma unroll
      for (int r = 0; r < 8; ++r) {
        int e = ebase + hi + r;
        atomicAdd(&agg[(long)dst[e] * HID + col], (acc2[ct][r] + b2) * wrow[r]);
      }
    }
  } else {
#pragma unroll
    for (int ct = 0; ct < 8; ++ct) {
      int col = ct * 16 + cl;
      float b2 = mb2[col];
#pragma unroll
      for (int r = 0; r < 8; ++r) {
        int e = ebase + hi + r;
        if (e < E)
          atomicAdd(&agg[(long)dst[e] * HID + col], (acc2[ct][r] + b2) * wrow[r]);
      }
    }
  }
}

// ---------------------------------------------------------------------------
// Kernel 5: update MLP + residual + layernorm
// ---------------------------------------------------------------------------
__global__ void k_update(const float* __restrict__ h, const float* __restrict__ agg,
                         const __bf16* __restrict__ Pw1, const float* __restrict__ ub1,
                         const __bf16* __restrict__ Pw2, const float* __restrict__ ub2,
                         const float* __restrict__ lng, const float* __restrict__ lnb,
                         float* __restrict__ hout, int N) {
  __shared__ float sin_[2][16][260];
  __shared__ float su[2][16][132];
  int lane = threadIdx.x & 31, wv = threadIdx.x >> 5;
  int nbase = (blockIdx.x * 2 + wv) * 16;
  float* in0 = &sin_[wv][0][0];
  float* u0 = &su[wv][0][0];
  for (int i = 0; i < 16; ++i) {
    int n = nbase + i; if (n >= N) n = N - 1;
    float* row = in0 + i * 260;
    async_row512(h + (long)n * HID, row);
    async_row512(agg + (long)n * HID, row + 128);
  }
  wait_async0();
  __syncthreads();
  int cl = lane & 15, hi = (lane < 16) ? 0 : 8;
  v8f acc1[8] = {};
  for (int kc = 0; kc < 8; ++kc) {
    v16bf a = load_A(in0, 260, kc * 32);
#pragma unroll
    for (int ct = 0; ct < 8; ++ct)
      acc1[ct] = wmma_bf16(a, load_Bp(Pw1, kc * 8 + ct), acc1[ct]);
  }
#pragma unroll
  for (int ct = 0; ct < 8; ++ct) {
    int col = ct * 16 + cl;
    float b1 = ub1[col];
#pragma unroll
    for (int r = 0; r < 8; ++r) {
      float v = acc1[ct][r] + b1;
      u0[(hi + r) * 132 + col] = v > 0.f ? v : 0.f;
    }
  }
  __syncthreads();
  v8f acc2[8] = {};
  for (int kc = 0; kc < 4; ++kc) {
    v16bf a = load_A(u0, 132, kc * 32);
#pragma unroll
    for (int ct = 0; ct < 8; ++ct)
      acc2[ct] = wmma_bf16(a, load_Bp(Pw2, kc * 8 + ct), acc2[ct]);
  }
  __syncthreads();                       // u0 consumed; reuse for pre-LN values
#pragma unroll
  for (int ct = 0; ct < 8; ++ct) {
    int col = ct * 16 + cl;
    float b2 = ub2[col];
#pragma unroll
    for (int r = 0; r < 8; ++r) {
      float pre = acc2[ct][r] + b2 + in0[(hi + r) * 260 + col];  // residual h
      u0[(hi + r) * 132 + col] = pre > 0.f ? pre : 0.f;
    }
  }
  __syncthreads();
  if (lane < 16) {
    int n = nbase + lane;
    if (n < N) {
      const float* row = u0 + lane * 132;
      float mu = 0.f;
      for (int j = 0; j < HID; ++j) mu += row[j];
      mu *= (1.0f / HID);
      float var = 0.f;
      for (int j = 0; j < HID; ++j) { float d = row[j] - mu; var += d * d; }
      var *= (1.0f / HID);
      float inv = rsqrtf(var + 1e-5f);
      float* op = hout + (long)n * HID;
      for (int j = 0; j < HID; ++j)
        op[j] = (row[j] - mu) * inv * lng[j] + lnb[j];
    }
  }
}

// ---------------------------------------------------------------------------
// Kernel 6: readout  q = relu([h|ctx] @ qw1^T + qb1) @ qw2^T + qb2
// ---------------------------------------------------------------------------
__global__ void k_readout(const float* __restrict__ h, const float* __restrict__ ctx,
                          const __bf16* __restrict__ Pw1, const float* __restrict__ qb1,
                          const float* __restrict__ qw2, const float* __restrict__ qb2,
                          float* __restrict__ out, int N) {
  __shared__ float sin_[2][16][196];
  int lane = threadIdx.x & 31, wv = threadIdx.x >> 5;
  int nbase = (blockIdx.x * 2 + wv) * 16;
  float* in0 = &sin_[wv][0][0];
  for (int i = 0; i < 16; ++i) {
    int n = nbase + i; if (n >= N) n = N - 1;
    float* row = in0 + i * 196;
    async_row512(h + (long)n * HID, row);
#pragma unroll
    for (int j = 0; j < 2; ++j) row[128 + j * 32 + lane] = ctx[j * 32 + lane];
  }
  wait_async0();
  __syncthreads();
  v8f acc[8] = {};
  for (int kc = 0; kc < 6; ++kc) {
    v16bf a = load_A(in0, 196, kc * 32);
#pragma unroll
    for (int ct = 0; ct < 8; ++ct)
      acc[ct] = wmma_bf16(a, load_Bp(Pw1, kc * 8 + ct), acc[ct]);
  }
  int cl = lane & 15, hi = (lane < 16) ? 0 : 8;
  float part[8];
#pragma unroll
  for (int r = 0; r < 8; ++r) part[r] = 0.f;
#pragma unroll
  for (int ct = 0; ct < 8; ++ct) {
    int j = ct * 16 + cl;
    float b1 = qb1[j], w2 = qw2[j];
#pragma unroll
    for (int r = 0; r < 8; ++r) {
      float v = acc[ct][r] + b1;
      v = v > 0.f ? v : 0.f;
      part[r] += v * w2;
    }
  }
#pragma unroll
  for (int r = 0; r < 8; ++r) {
    part[r] += __shfl_xor(part[r], 1);
    part[r] += __shfl_xor(part[r], 2);
    part[r] += __shfl_xor(part[r], 4);
    part[r] += __shfl_xor(part[r], 8);
  }
  float b2 = qb2[0];
  if (cl == 0) {
    if (nbase + 16 <= N) {
#pragma unroll
      for (int r = 0; r < 8; ++r) out[nbase + hi + r] = part[r] + b2;
    } else {
#pragma unroll
      for (int r = 0; r < 8; ++r) {
        int n = nbase + hi + r;
        if (n < N) out[n] = part[r] + b2;
      }
    }
  }
}

// ---------------------------------------------------------------------------
// Host launcher
// ---------------------------------------------------------------------------
extern "C" void kernel_launch(void* const* d_in, const int* in_sizes, int n_in,
                              void* d_out, int out_size, void* d_ws, size_t ws_size,
                              hipStream_t stream) {
  const float* x    = (const float*)d_in[0];
  const int*   ei   = (const int*)d_in[1];
  const float* ea   = (const float*)d_in[2];
  const float* gctx = (const float*)d_in[3];
  const float* npw  = (const float*)d_in[4];
  const float* npb  = (const float*)d_in[5];
  const float* mw1  = (const float*)d_in[6];
  const float* mb1  = (const float*)d_in[7];
  const float* mw2  = (const float*)d_in[8];
  const float* mb2  = (const float*)d_in[9];
  const float* aw1  = (const float*)d_in[10];
  const float* ab1  = (const float*)d_in[11];
  const float* aw2  = (const float*)d_in[12];
  const float* ab2  = (const float*)d_in[13];
  const float* uw1  = (const float*)d_in[14];
  const float* ub1  = (const float*)d_in[15];
  const float* uw2  = (const float*)d_in[16];
  const float* ub2  = (const float*)d_in[17];
  const float* lng  = (const float*)d_in[18];
  const float* lnb  = (const float*)d_in[19];
  const float* qw1  = (const float*)d_in[20];
  const float* qb1  = (const float*)d_in[21];
  const float* qw2  = (const float*)d_in[22];
  const float* qb2  = (const float*)d_in[23];
  float* out = (float*)d_out;

  const int N = in_sizes[0] / 32;
  const int E = in_sizes[2] / 16;
  const int* src = ei;
  const int* dst = ei + E;

  // ---- workspace: pre-packed bf16 weight tiles first (32B-aligned blocks) --
  __bf16* pk = (__bf16*)d_ws;
  __bf16* Pnp = pk;                 pk += 8  * 512;   // ct8, kc1
  __bf16* Pq  = pk;                 pk += 48 * 512;   // ct8, kc6
  __bf16 *Paw1[3], *Pmw1[3], *Pmw2[3], *Puw1[3], *Puw2[3];
  for (int l = 0; l < 3; ++l) {
    Paw1[l] = pk; pk += 36 * 512;   // ct4, kc9
    Pmw1[l] = pk; pk += 40 * 512;   // ct8, kc5
    Pmw2[l] = pk; pk += 32 * 512;   // ct8, kc4
    Puw1[l] = pk; pk += 64 * 512;   // ct8, kc8
    Puw2[l] = pk; pk += 32 * 512;   // ct8, kc4
  }
  float* hA      = (float*)pk;
  float* hB      = hA + (size_t)N * HID;
  float* agg     = hB + (size_t)N * HID;
  float* scores  = agg + (size_t)N * HID;
  float* expsc   = scores + E;
  float* sum_exp = expsc + E;
  unsigned* maxbits = (unsigned*)(sum_exp + N);

  auto pack = [&](const float* W, __bf16* P, int ld, int kmax, int ctn, int kcn) {
    int total = ctn * kcn * 512;
    k_pack<<<(total + 255) / 256, 256, 0, stream>>>(W, P, ld, kmax, ctn, total);
  };
  pack(npw, Pnp, 32, 32, 8, 1);
  pack(qw1, Pq, 192, 192, 8, 6);
  for (int l = 0; l < 3; ++l) {
    pack(aw1 + (size_t)l * 64 * 272,  Paw1[l], 272, 272, 4, 9);
    pack(mw1 + (size_t)l * 128 * 144, Pmw1[l], 144, 144, 8, 5);
    pack(mw2 + (size_t)l * 128 * 128, Pmw2[l], 128, 128, 8, 4);
    pack(uw1 + (size_t)l * 128 * 256, Puw1[l], 256, 256, 8, 8);
    pack(uw2 + (size_t)l * 128 * 128, Puw2[l], 128, 128, 8, 4);
  }

  const int nblk = (N + 31) / 32;   // 2 waves/block, 16 rows/wave
  const int eblk = (E + 31) / 32;

  k_node_proj<<<nblk, 64, 0, stream>>>(x, Pnp, npb, hA, N);

  const float* hin = hA;
  float* hout = hB;
  for (int l = 0; l < 3; ++l) {
    hipMemsetAsync(maxbits, 0, sizeof(unsigned), stream);
    hipMemsetAsync(sum_exp, 0, (size_t)N * sizeof(float), stream);
    hipMemsetAsync(agg, 0, (size_t)N * HID * sizeof(float), stream);

    k_attn<<<eblk, 64, 0, stream>>>(hin, ea, src, dst,
                                    Paw1[l], ab1 + (size_t)l * 64,
                                    aw2 + (size_t)l * 64, ab2 + l,
                                    scores, maxbits, E);
    k_expsum<<<(E + 255) / 256, 256, 0, stream>>>(scores, dst, maxbits,
                                                  expsc, sum_exp, E);
    k_msg<<<eblk, 64, 0, stream>>>(hin, ea, src, dst,
                                   Pmw1[l], mb1 + (size_t)l * 128,
                                   Pmw2[l], mb2 + (size_t)l * 128,
                                   expsc, sum_exp, agg, E);
    k_update<<<nblk, 64, 0, stream>>>(hin, agg,
                                      Puw1[l], ub1 + (size_t)l * 128,
                                      Puw2[l], ub2 + (size_t)l * 128,
                                      lng + (size_t)l * 128, lnb + (size_t)l * 128,
                                      hout, N);
    const float* t = hin; hin = hout; hout = (float*)t;
  }

  k_readout<<<nblk, 64, 0, stream>>>(hin, gctx, Pq, qb1, qw2, qb2, out, N);
}
